// MultiRPN_78778290144146
// MI455X (gfx1250) — compile-verified, matrix operand order checked by reading, not run on hardware
//
#include <hip/hip_runtime.h>

typedef _Float16 half_t;
typedef __attribute__((ext_vector_type(16))) _Float16 v16h;
typedef __attribute__((ext_vector_type(8)))  _Float16 v8h;
typedef __attribute__((ext_vector_type(8)))  float    v8f;

static constexpr int S  = 3;
static constexpr int B  = 64;
static constexpr int C  = 256;
static constexpr int HD = 256;
static constexpr int XS = 31, XO = 29;   // search in/out spatial
static constexpr int ZS = 7,  ZO = 5;    // template in/out spatial
static constexpr int P  = 25;            // xcorr output spatial
static constexpr int M  = B * P * P;     // 40000 GEMM rows (valid)
static constexpr int Mp = 40064;         // padded to multiple of 128
static constexpr float EPS = 1e-5f;

// ---------------- small prep kernels ----------------

__global__ void f32_to_f16_kernel(const float* __restrict__ src, half_t* __restrict__ dst, int n) {
    int i = blockIdx.x * blockDim.x + threadIdx.x;
    if (i < n) dst[i] = (half_t)src[i];
}

__global__ void bn_prep_kernel(const float* __restrict__ g, const float* __restrict__ b,
                               const float* __restrict__ m, const float* __restrict__ v,
                               float* __restrict__ scale, float* __restrict__ bias, int n) {
    int i = blockIdx.x * blockDim.x + threadIdx.x;
    if (i < n) {
        float inv = g[i] * rsqrtf(v[i] + EPS);
        scale[i] = inv;
        bias[i]  = b[i] - m[i] * inv;
    }
}

// ---------------- depthwise 3x3 + BN + ReLU (f32 in, f16 out) ----------------

__global__ void dw3_bn_relu_kernel(const float* __restrict__ x, const float* __restrict__ w,
                                   const float* __restrict__ g, const float* __restrict__ bb,
                                   const float* __restrict__ mm, const float* __restrict__ vv,
                                   half_t* __restrict__ out,
                                   int H, int Wd, int OH, int OW, int total) {
    int i = blockIdx.x * blockDim.x + threadIdx.x;
    if (i >= total) return;
    int ow = i % OW;
    int t  = i / OW;
    int oh = t % OH; t /= OH;
    int c  = t % C;
    int b  = t / C;
    const float* xp = x + ((size_t)(b * C + c) * H + oh) * Wd + ow;
    const float* wp = w + c * 9;
    float s = 0.f;
#pragma unroll
    for (int ky = 0; ky < 3; ++ky)
#pragma unroll
        for (int kx = 0; kx < 3; ++kx)
            s += xp[ky * Wd + kx] * wp[ky * 3 + kx];
    float inv = g[c] * rsqrtf(vv[c] + EPS);
    float val = s * inv + (bb[c] - mm[c] * inv);
    out[i] = (half_t)fmaxf(val, 0.f);
}

// ---------------- depthwise 5x5 cross-correlation, writes GEMM-A layout [M][C] ----------------

__global__ void xcorr_kernel(const half_t* __restrict__ search, const half_t* __restrict__ kern,
                             half_t* __restrict__ A1, int total) {
    int i = blockIdx.x * blockDim.x + threadIdx.x;
    if (i >= total) return;
    int c = i % C;
    int t = i / C;
    int p = t % (P * P);
    int b = t / (P * P);
    int py = p / P, px = p % P;
    const half_t* sp = search + ((size_t)(b * C + c) * XO + py) * XO + px;
    const half_t* kp = kern + (size_t)(b * C + c) * (ZO * ZO);
    float acc = 0.f;
#pragma unroll
    for (int ky = 0; ky < ZO; ++ky)
#pragma unroll
        for (int kx = 0; kx < ZO; ++kx)
            acc += (float)sp[ky * XO + kx] * (float)kp[ky * ZO + kx];
    A1[(size_t)(b * (P * P) + p) * C + c] = (half_t)acc;
}

// ---------------- WMMA GEMM: Out = relu(bn(A[Mp][256] x W[256][256]^T)), f16 out ----------------
// Block = 256 threads = 8 waves, arranged 4 (M) x 2 (N). Wave tile = 32x64 = 2x4 WMMAs.
// A fragment (16x32 f16, per ISA layout): lane L holds row M = L%16; element j holds
// K = kb*32 + (L>=16 ? 8 : 0) + (j<8 ? j : j+8)  -> two contiguous 16B chunks (one b128 pair).
// B fragment: lane L holds column n = L%16 of B == row n of W (W is [N][K] row-major),
// same K indexing -> same load pattern from W.
// All Out buffers have Mp (padded) rows and pad rows are never consumed, so the
// epilogue stores unconditionally (no per-element exec-mask branching).

__global__ __launch_bounds__(256)
void wmma_gemm_bn_relu_kernel(const half_t* __restrict__ A, const half_t* __restrict__ W,
                              const float* __restrict__ scale, const float* __restrict__ bias,
                              half_t* __restrict__ Out) {
    const int K = 256;
    int lane = threadIdx.x & 31;
    int wave = threadIdx.x >> 5;
    int wm = wave & 3;        // 4 waves along M
    int wn = wave >> 2;       // 2 waves along N
    int lr  = lane & 15;
    int lhi = lane >> 4;
    int blockM = blockIdx.x * 128;
    int blockN = blockIdx.y * 128;
    int mrow0 = blockM + wm * 32 + lr;   // row loaded by this lane (A frags)
    int nrow0 = blockN + wn * 64 + lr;   // W row loaded by this lane (B frags)

    v8f acc[2][4] = {};

    for (int kb = 0; kb < 8; ++kb) {
        int kbase = kb * 32 + lhi * 8;
        v16h a[2], bf[4];
#pragma unroll
        for (int i = 0; i < 2; ++i) {
            const half_t* ap = A + (size_t)(mrow0 + i * 16) * K + kbase;
            v8h lo = *(const v8h*)ap;
            v8h hi = *(const v8h*)(ap + 16);
            a[i] = __builtin_shufflevector(lo, hi, 0, 1, 2, 3, 4, 5, 6, 7,
                                           8, 9, 10, 11, 12, 13, 14, 15);
        }
#pragma unroll
        for (int j = 0; j < 4; ++j) {
            const half_t* wp = W + (size_t)(nrow0 + j * 16) * K + kbase;
            v8h lo = *(const v8h*)wp;
            v8h hi = *(const v8h*)(wp + 16);
            bf[j] = __builtin_shufflevector(lo, hi, 0, 1, 2, 3, 4, 5, 6, 7,
                                            8, 9, 10, 11, 12, 13, 14, 15);
        }
#pragma unroll
        for (int i = 0; i < 2; ++i)
#pragma unroll
            for (int j = 0; j < 4; ++j)
                acc[i][j] = __builtin_amdgcn_wmma_f32_16x16x32_f16(
                    false, a[i], false, bf[j], (short)0, acc[i][j], false, false);
    }

    // Epilogue: C/D layout -> lane holds col n = tile + lr, rows tile + r + lhi*8.
#pragma unroll
    for (int j = 0; j < 4; ++j) {
        int n = blockN + wn * 64 + j * 16 + lr;
        float sc = scale[n];
        float bi = bias[n];
#pragma unroll
        for (int i = 0; i < 2; ++i) {
            int mbase = blockM + wm * 32 + i * 16 + lhi * 8;
#pragma unroll
            for (int r = 0; r < 8; ++r) {
                int m = mbase + r;
                float v = fmaf(acc[i][j][r], sc, bi);
                Out[(size_t)m * K + n] = (half_t)fmaxf(v, 0.f);
            }
        }
    }
}

// ---------------- final: w2 projection + b2 + softmax-weighted accumulation over scales ----------------

__global__ void head_out_kernel(const half_t* __restrict__ h_cls, const half_t* __restrict__ h_loc,
                                const half_t* __restrict__ h_ctr,
                                const float* __restrict__ cls_w2, const float* __restrict__ cls_b2,
                                const float* __restrict__ loc_w2, const float* __restrict__ loc_b2,
                                const float* __restrict__ ctr_w2, const float* __restrict__ ctr_b2,
                                const float* __restrict__ w_cls, const float* __restrict__ w_loc,
                                const float* __restrict__ w_ctr,
                                float* __restrict__ out, int s) {
    int i = blockIdx.x * blockDim.x + threadIdx.x;
    if (i >= 7 * M) return;
    int o  = i / M;
    int mp = i % M;
    int b  = mp / (P * P);
    int p  = mp % (P * P);

    const half_t* h; const float* w2p; const float* b2p; const float* wv;
    int oo, oidx;
    if (o < 2)      { h = h_cls; w2p = cls_w2; b2p = cls_b2; wv = w_cls; oo = o;
                      oidx = (b * 2 + oo) * (P * P) + p; }
    else if (o < 6) { h = h_loc; w2p = loc_w2; b2p = loc_b2; wv = w_loc; oo = o - 2;
                      oidx = 2 * B * P * P + (b * 4 + oo) * (P * P) + p; }
    else            { h = h_ctr; w2p = ctr_w2; b2p = ctr_b2; wv = w_ctr; oo = 0;
                      oidx = 6 * B * P * P + b * (P * P) + p; }

    float wa = wv[0], wb = wv[1], wc = wv[2];
    float mx = fmaxf(wa, fmaxf(wb, wc));
    float e0 = __expf(wa - mx), e1 = __expf(wb - mx), e2 = __expf(wc - mx);
    float es = (s == 0) ? e0 : (s == 1) ? e1 : e2;
    float wt = es / (e0 + e1 + e2);

    // vectorized 256-long dot product: 32 x v8h (16B) activation loads
    const v8h*  hv = (const v8h*)(h + (size_t)mp * HD);
    const float* wp = w2p + oo * HD;
    float acc = 0.f;
#pragma unroll 8
    for (int k = 0; k < HD / 8; ++k) {
        v8h hx = hv[k];
        const float* wk = wp + k * 8;
#pragma unroll
        for (int e = 0; e < 8; ++e)
            acc = fmaf((float)hx[e], wk[e], acc);
    }
    float val = (acc + b2p[oo]) * wt;

    if (s == 0) out[oidx] = val;
    else        out[oidx] += val;
}

// ---------------- host ----------------

extern "C" void kernel_launch(void* const* d_in, const int* in_sizes, int n_in,
                              void* d_out, int out_size, void* d_ws, size_t ws_size,
                              hipStream_t stream) {
    (void)in_sizes; (void)n_in; (void)out_size; (void)ws_size;

    const float* z_fs   = (const float*)d_in[0];
    const float* x_fs   = (const float*)d_in[1];
    const float* pre_w  = (const float*)d_in[2];
    const float* pre_g  = (const float*)d_in[3];
    const float* pre_b  = (const float*)d_in[4];
    const float* pre_m  = (const float*)d_in[5];
    const float* pre_v  = (const float*)d_in[6];
    const float* head_w = (const float*)d_in[7];
    const float* head_g = (const float*)d_in[8];
    const float* head_b = (const float*)d_in[9];
    const float* head_m = (const float*)d_in[10];
    const float* head_v = (const float*)d_in[11];
    const float* cls_w1 = (const float*)d_in[12];
    const float* cls_g  = (const float*)d_in[13];
    const float* cls_b  = (const float*)d_in[14];
    const float* cls_m  = (const float*)d_in[15];
    const float* cls_v  = (const float*)d_in[16];
    const float* cls_w2 = (const float*)d_in[17];
    const float* cls_b2 = (const float*)d_in[18];
    const float* loc_w1 = (const float*)d_in[19];
    const float* loc_g  = (const float*)d_in[20];
    const float* loc_b  = (const float*)d_in[21];
    const float* loc_m  = (const float*)d_in[22];
    const float* loc_v  = (const float*)d_in[23];
    const float* loc_w2 = (const float*)d_in[24];
    const float* loc_b2 = (const float*)d_in[25];
    const float* ctr_w1 = (const float*)d_in[26];
    const float* ctr_g  = (const float*)d_in[27];
    const float* ctr_b  = (const float*)d_in[28];
    const float* ctr_m  = (const float*)d_in[29];
    const float* ctr_v  = (const float*)d_in[30];
    const float* ctr_w2 = (const float*)d_in[31];
    const float* ctr_b2 = (const float*)d_in[32];
    const float* w_cls  = (const float*)d_in[33];
    const float* w_loc  = (const float*)d_in[34];
    const float* w_ctr  = (const float*)d_in[35];

    char* wsp = (char*)d_ws;
    auto alloc = [&](size_t bytes) {
        char* p = wsp;
        wsp += (bytes + 255) & ~(size_t)255;
        return p;
    };

    // per-scale reusable buffers (~132 MB total)
    half_t* search = (half_t*)alloc((size_t)B * C * XO * XO * sizeof(half_t));
    half_t* kern   = (half_t*)alloc((size_t)B * C * ZO * ZO * sizeof(half_t));
    half_t* A1     = (half_t*)alloc((size_t)Mp * C  * sizeof(half_t));
    half_t* feat   = (half_t*)alloc((size_t)Mp * HD * sizeof(half_t));
    half_t* hc     = (half_t*)alloc((size_t)Mp * HD * sizeof(half_t));
    half_t* hl     = (half_t*)alloc((size_t)Mp * HD * sizeof(half_t));
    half_t* ht     = (half_t*)alloc((size_t)Mp * HD * sizeof(half_t));
    half_t* Wh     = (half_t*)alloc((size_t)4 * HD * C * sizeof(half_t)); // head, cls, loc, ctr
    float*  sb     = (float*) alloc((size_t)4 * 2 * HD * sizeof(float));  // scale+bias per set

    half_t* Wh_head = Wh;
    half_t* Wh_cls  = Wh + 1 * HD * C;
    half_t* Wh_loc  = Wh + 2 * HD * C;
    half_t* Wh_ctr  = Wh + 3 * HD * C;
    float* sc_head = sb + 0 * HD; float* bi_head = sb + 1 * HD;
    float* sc_cls  = sb + 2 * HD; float* bi_cls  = sb + 3 * HD;
    float* sc_loc  = sb + 4 * HD; float* bi_loc  = sb + 5 * HD;
    float* sc_ctr  = sb + 6 * HD; float* bi_ctr  = sb + 7 * HD;

    const int WN = HD * C;               // 65536 weights per matrix
    dim3 gemm_grid(Mp / 128, HD / 128);  // (313, 2)

    for (int s = 0; s < S; ++s) {
        const float* xs  = x_fs + (size_t)s * B * C * XS * XS;
        const float* zs  = z_fs + (size_t)s * B * C * ZS * ZS;
        const float* prw = pre_w + (size_t)s * C * 9;

        // weights -> f16, BN folding
        f32_to_f16_kernel<<<(WN + 255) / 256, 256, 0, stream>>>(head_w + (size_t)s * WN, Wh_head, WN);
        f32_to_f16_kernel<<<(WN + 255) / 256, 256, 0, stream>>>(cls_w1 + (size_t)s * WN, Wh_cls, WN);
        f32_to_f16_kernel<<<(WN + 255) / 256, 256, 0, stream>>>(loc_w1 + (size_t)s * WN, Wh_loc, WN);
        f32_to_f16_kernel<<<(WN + 255) / 256, 256, 0, stream>>>(ctr_w1 + (size_t)s * WN, Wh_ctr, WN);
        bn_prep_kernel<<<1, 256, 0, stream>>>(head_g + s * HD, head_b + s * HD, head_m + s * HD, head_v + s * HD, sc_head, bi_head, HD);
        bn_prep_kernel<<<1, 256, 0, stream>>>(cls_g + s * HD, cls_b + s * HD, cls_m + s * HD, cls_v + s * HD, sc_cls, bi_cls, HD);
        bn_prep_kernel<<<1, 256, 0, stream>>>(loc_g + s * HD, loc_b + s * HD, loc_m + s * HD, loc_v + s * HD, sc_loc, bi_loc, HD);
        bn_prep_kernel<<<1, 256, 0, stream>>>(ctr_g + s * HD, ctr_b + s * HD, ctr_m + s * HD, ctr_v + s * HD, sc_ctr, bi_ctr, HD);

        // depthwise 3x3 + BN + ReLU
        int tot_x = B * C * XO * XO;
        int tot_z = B * C * ZO * ZO;
        dw3_bn_relu_kernel<<<(tot_x + 255) / 256, 256, 0, stream>>>(
            xs, prw, pre_g + s * C, pre_b + s * C, pre_m + s * C, pre_v + s * C,
            search, XS, XS, XO, XO, tot_x);
        dw3_bn_relu_kernel<<<(tot_z + 255) / 256, 256, 0, stream>>>(
            zs, prw, pre_g + s * C, pre_b + s * C, pre_m + s * C, pre_v + s * C,
            kern, ZS, ZS, ZO, ZO, tot_z);

        // depthwise cross-correlation -> A1 [M][C]
        int tot_xc = B * P * P * C;
        xcorr_kernel<<<(tot_xc + 255) / 256, 256, 0, stream>>>(search, kern, A1, tot_xc);

        // WMMA GEMM chain
        wmma_gemm_bn_relu_kernel<<<gemm_grid, 256, 0, stream>>>(A1,   Wh_head, sc_head, bi_head, feat);
        wmma_gemm_bn_relu_kernel<<<gemm_grid, 256, 0, stream>>>(feat, Wh_cls,  sc_cls,  bi_cls,  hc);
        wmma_gemm_bn_relu_kernel<<<gemm_grid, 256, 0, stream>>>(feat, Wh_loc,  sc_loc,  bi_loc,  hl);
        wmma_gemm_bn_relu_kernel<<<gemm_grid, 256, 0, stream>>>(feat, Wh_ctr,  sc_ctr,  bi_ctr,  ht);

        // final projection + weighted accumulation
        head_out_kernel<<<(7 * M + 255) / 256, 256, 0, stream>>>(
            hc, hl, ht,
            cls_w2 + (size_t)s * 2 * HD, cls_b2 + (size_t)s * 2,
            loc_w2 + (size_t)s * 4 * HD, loc_b2 + (size_t)s * 4,
            ctr_w2 + (size_t)s * 1 * HD, ctr_b2 + (size_t)s * 1,
            w_cls, w_loc, w_ctr, (float*)d_out, s);
    }
}
